// WSVectorQuantizer_61787399520296
// MI455X (gfx1250) — matrix operand order repaired and checked by reading, not compiled, not run on hardware
//
#include <hip/hip_runtime.h>
#include <math.h>

typedef __bf16 bf16_t;
typedef __attribute__((ext_vector_type(8)))  __bf16 v8bf;
typedef __attribute__((ext_vector_type(16))) __bf16 v16bf;
typedef __attribute__((ext_vector_type(8)))  float  v8f;

#define D       256
#define ROWS_PB 128   // rows per block (8 waves x 16 rows)
#define COLS_PS 32    // codebook columns staged per step (2 x 16 per wave)

// ---------------------------------------------------------------------------
// Pass 0: split codebook into bf16 hi/lo planes, compute ||c_k||^2 (exact f32),
// zero the histogram. One block per codebook row; fully coalesced.
// ---------------------------------------------------------------------------
__global__ __launch_bounds__(256) void vq_prep_kernel(
    const float* __restrict__ cb, float* __restrict__ cnorm,
    float* __restrict__ counts, bf16_t* __restrict__ cbHi,
    bf16_t* __restrict__ cbLo) {
  const int k = blockIdx.x;
  const int d = threadIdx.x;
  const float x = cb[(size_t)k * D + d];
  const bf16_t hi = (bf16_t)x;                  // RNE
  const bf16_t lo = (bf16_t)(x - (float)hi);    // exact residual, then RNE
  cbHi[(size_t)k * D + d] = hi;
  cbLo[(size_t)k * D + d] = lo;

  __shared__ float sb[256];
  sb[d] = x * x;
  __syncthreads();
  for (int off = 128; off > 0; off >>= 1) {
    if (d < off) sb[d] += sb[d + off];
    __syncthreads();
  }
  if (d == 0) { cnorm[k] = sb[0]; counts[k] = 0.f; }
}

// Build a 16-element bf16 fragment from two aligned 8-element LDS runs.
__device__ __forceinline__ v16bf ldfrag(const bf16_t* p0, const bf16_t* p1) {
  v8bf a = *(const v8bf*)p0;
  v8bf b = *(const v8bf*)p1;
  return __builtin_shufflevector(a, b, 0, 1, 2, 3, 4, 5, 6, 7,
                                       8, 9, 10, 11, 12, 13, 14, 15);
}

// ---------------------------------------------------------------------------
// Pass 1: 256 threads = 8 waves; block owns 128 rows of z. Each wave computes
// a 16x32 tile per step (two accumulators, A fragments reused across both
// column tiles) with bf16x3 WMMA. B tiles are ping-pong double-buffered in
// LDS with register prefetch so global latency hides under the WMMA loop.
// ---------------------------------------------------------------------------
__global__ __launch_bounds__(256) void vq_argmin_gather_kernel(
    const float* __restrict__ z, const float* __restrict__ codebook,
    const bf16_t* __restrict__ cbHi, const bf16_t* __restrict__ cbLo,
    const float* __restrict__ cnorm, float* __restrict__ counts,
    float* __restrict__ zq, int K) {
  __shared__ bf16_t Ahi[ROWS_PB * D];        // 64 KB
  __shared__ bf16_t Alo[ROWS_PB * D];        // 64 KB
  __shared__ bf16_t Bhi[2][COLS_PS * D];     // 32 KB ping-pong
  __shared__ bf16_t Blo[2][COLS_PS * D];     // 32 KB ping-pong
  __shared__ float  redVal[8][16][16];       //  8 KB
  __shared__ int    redIdx[8][16][16];       //  8 KB
  __shared__ int    finalIdx[ROWS_PB];

  const int tid  = threadIdx.x;
  const int lane = tid & 31;
  const int wid  = tid >> 5;
  const int m16  = lane & 15;
  const int half = lane >> 4;
  const size_t rowbase = (size_t)blockIdx.x * ROWS_PB;
  const int NSTEP = K / COLS_PS;             // 32

  // Stage + hi/lo split of the 128x256 z tile (coalesced float4 loads).
  {
    const float4* src = (const float4*)(z + rowbase * D);
    for (int i = tid; i < ROWS_PB * D / 4; i += 256) {
      float4 v = src[i];
      int b = i * 4;
      bf16_t hx = (bf16_t)v.x, hy = (bf16_t)v.y, hz = (bf16_t)v.z, hw = (bf16_t)v.w;
      Ahi[b + 0] = hx; Ahi[b + 1] = hy; Ahi[b + 2] = hz; Ahi[b + 3] = hw;
      Alo[b + 0] = (bf16_t)(v.x - (float)hx);
      Alo[b + 1] = (bf16_t)(v.y - (float)hy);
      Alo[b + 2] = (bf16_t)(v.z - (float)hz);
      Alo[b + 3] = (bf16_t)(v.w - (float)hw);
    }
  }

  float mval[8];
  int   midx[8];
#pragma unroll
  for (int r = 0; r < 8; ++r) { mval[r] = 3.4e38f; midx[r] = 0; }

  // A: lane m16 = row M; half selects K run {half*16..+15}.
  const bf16_t* aHiRow = Ahi + ((size_t)(wid * 16 + m16)) * D;
  const bf16_t* aLoRow = Alo + ((size_t)(wid * 16 + m16)) * D;

  // --- B tile staging helpers (32 cols x 256 d, hi+lo = 2 x 16 KB/step) ---
  const uint4* gHi = (const uint4*)cbHi;   // 1024 uint4 per step per plane
  const uint4* gLo = (const uint4*)cbLo;
  uint4 rh[4], rl[4];

  // Prologue: fetch step 0, park in buffer 0.
#pragma unroll
  for (int i = 0; i < 4; ++i) {
    rh[i] = gHi[tid + 256 * i];
    rl[i] = gLo[tid + 256 * i];
  }
  {
    uint4* dh = (uint4*)Bhi[0];
    uint4* dl = (uint4*)Blo[0];
#pragma unroll
    for (int i = 0; i < 4; ++i) { dh[tid + 256 * i] = rh[i]; dl[tid + 256 * i] = rl[i]; }
  }
  __syncthreads();

  for (int s = 0; s < NSTEP; ++s) {
    const int p = s & 1;
    // Issue next tile's global loads now; consumed after the WMMA loop.
    if (s + 1 < NSTEP) {
      const size_t off = (size_t)(s + 1) * (COLS_PS * D / 8);  // uint4 units
#pragma unroll
      for (int i = 0; i < 4; ++i) {
        rh[i] = gHi[off + tid + 256 * i];
        rl[i] = gLo[off + tid + 256 * i];
      }
    }

    // B: lane m16 = column N (tile0) / 16+m16 (tile1); K run = d0+half*16..+15.
    const bf16_t* b0Hi = Bhi[p] + (size_t)m16 * D;
    const bf16_t* b0Lo = Blo[p] + (size_t)m16 * D;
    const bf16_t* b1Hi = Bhi[p] + (size_t)(16 + m16) * D;
    const bf16_t* b1Lo = Blo[p] + (size_t)(16 + m16) * D;

    v8f acc0 = {};
    v8f acc1 = {};
#pragma unroll
    for (int d0 = 0; d0 < D; d0 += 32) {
      const int ra = d0 + half * 16;
      const int rb = d0 + half * 16 + 8;
      const int rc = d0 + half * 16;
      v16bf ah  = ldfrag(aHiRow + ra, aHiRow + rb);
      v16bf al  = ldfrag(aLoRow + ra, aLoRow + rb);
      v16bf bh0 = ldfrag(b0Hi + rc, b0Hi + rc + 8);
      v16bf bl0 = ldfrag(b0Lo + rc, b0Lo + rc + 8);
      v16bf bh1 = ldfrag(b1Hi + rc, b1Hi + rc + 8);
      v16bf bl1 = ldfrag(b1Lo + rc, b1Lo + rc + 8);
      acc0 = __builtin_amdgcn_wmma_f32_16x16x32_bf16(false, ah, false, bh0,
                                                     (short)0, acc0, false, false);
      acc1 = __builtin_amdgcn_wmma_f32_16x16x32_bf16(false, ah, false, bh1,
                                                     (short)0, acc1, false, false);
      acc0 = __builtin_amdgcn_wmma_f32_16x16x32_bf16(false, ah, false, bl0,
                                                     (short)0, acc0, false, false);
      acc1 = __builtin_amdgcn_wmma_f32_16x16x32_bf16(false, ah, false, bl1,
                                                     (short)0, acc1, false, false);
      acc0 = __builtin_amdgcn_wmma_f32_16x16x32_bf16(false, al, false, bh0,
                                                     (short)0, acc0, false, false);
      acc1 = __builtin_amdgcn_wmma_f32_16x16x32_bf16(false, al, false, bh1,
                                                     (short)0, acc1, false, false);
    }

    // cost = ||c_k||^2 - 2*dot ; running argmin, ascending k preserves ties.
    {
      const int k0 = s * COLS_PS + m16;
      const float cn0 = cnorm[k0];
#pragma unroll
      for (int r = 0; r < 8; ++r) {
        float cost = cn0 - 2.0f * acc0[r];
        if (cost < mval[r]) { mval[r] = cost; midx[r] = k0; }
      }
      const int k1 = k0 + 16;
      const float cn1 = cnorm[k1];
#pragma unroll
      for (int r = 0; r < 8; ++r) {
        float cost = cn1 - 2.0f * acc1[r];
        if (cost < mval[r]) { mval[r] = cost; midx[r] = k1; }
      }
    }

    // Park the prefetched tile in the other buffer; readers of buf[p] are
    // untouched, so a single barrier orders store -> next compute.
    if (s + 1 < NSTEP) {
      uint4* dh = (uint4*)Bhi[p ^ 1];
      uint4* dl = (uint4*)Blo[p ^ 1];
#pragma unroll
      for (int i = 0; i < 4; ++i) { dh[tid + 256 * i] = rh[i]; dl[tid + 256 * i] = rl[i]; }
    }
    __syncthreads();
  }

  // Cross-lane argmin within each wave via LDS.
#pragma unroll
  for (int r = 0; r < 8; ++r) {
    int row = r + half * 8;
    redVal[wid][row][m16] = mval[r];
    redIdx[wid][row][m16] = midx[r];
  }
  __syncthreads();

  if (lane < 16) {
    float best = redVal[wid][lane][0];
    int   bi   = redIdx[wid][lane][0];
#pragma unroll
    for (int c = 1; c < 16; ++c) {
      float v = redVal[wid][lane][c];
      int  id = redIdx[wid][lane][c];
      if (v < best || (v == best && id < bi)) { best = v; bi = id; }
    }
    finalIdx[wid * 16 + lane] = bi;
    atomicAdd(&counts[bi], 1.0f);   // global_atomic_add_f32
  }
  __syncthreads();

  // z_q gather: 128 rows x 64 float4, flat over 256 threads (coalesced).
  for (int i = tid; i < ROWS_PB * (D / 4); i += 256) {
    int row = i >> 6;
    int col = i & 63;
    const float4* src = (const float4*)(codebook + (size_t)finalIdx[row] * D);
    float4* dst = (float4*)(zq + (rowbase + row) * D);
    dst[col] = src[col];
  }
}

// ---------------------------------------------------------------------------
// Pass 2: perplexity = exp(-sum(e*log(e+1e-10))), loss = 0 (eval path).
// ---------------------------------------------------------------------------
__global__ __launch_bounds__(256) void vq_perplexity_kernel(
    const float* __restrict__ counts, float* __restrict__ out,
    int K, float invN, size_t outOff) {
  __shared__ float sbuf[256];
  float s = 0.f;
  for (int k = threadIdx.x; k < K; k += 256) {
    float e = counts[k] * invN;
    s += e * logf(e + 1e-10f);
  }
  sbuf[threadIdx.x] = s;
  __syncthreads();
  for (int off = 128; off > 0; off >>= 1) {
    if ((int)threadIdx.x < off) sbuf[threadIdx.x] += sbuf[threadIdx.x + off];
    __syncthreads();
  }
  if (threadIdx.x == 0) {
    out[outOff]     = 0.0f;
    out[outOff + 1] = expf(-sbuf[0]);
  }
}

extern "C" void kernel_launch(void* const* d_in, const int* in_sizes, int n_in,
                              void* d_out, int out_size, void* d_ws, size_t ws_size,
                              hipStream_t stream) {
  const float* z        = (const float*)d_in[0];
  const float* codebook = (const float*)d_in[1];
  // d_in[2] = codebook_weight (unused in eval), d_in[3] = flg_train (0)
  (void)n_in; (void)out_size; (void)ws_size;

  const int N = in_sizes[0] / D;   // 65536
  const int K = in_sizes[1] / D;   // 1024

  // Workspace: cnorm[K] | counts[K] | cbHi[K*D] bf16 | cbLo[K*D] bf16  (~1 MB)
  float*  cnorm  = (float*)d_ws;
  float*  counts = cnorm + K;
  bf16_t* cbHi   = (bf16_t*)(counts + K);
  bf16_t* cbLo   = cbHi + (size_t)K * D;
  float*  out    = (float*)d_out;

  vq_prep_kernel<<<K, 256, 0, stream>>>(codebook, cnorm, counts, cbHi, cbLo);
  vq_argmin_gather_kernel<<<N / ROWS_PB, 256, 0, stream>>>(
      z, codebook, cbHi, cbLo, cnorm, counts, out, K);
  vq_perplexity_kernel<<<1, 256, 0, stream>>>(counts, out, K, 1.0f / (float)N,
                                              (size_t)N * D);
}
